// NonLocal_27900107554980
// MI455X (gfx1250) — matrix operand order repaired and checked by reading, not compiled, not run on hardware
//
#include <hip/hip_runtime.h>

typedef __attribute__((ext_vector_type(16))) __bf16 v16bf;
typedef __attribute__((ext_vector_type(8)))  float  v8f;

#define NTOK 16384
#define C_IN 128
#define CH   64      // C2 / head dim

struct u4x2 { uint4 a, b; };

// A fragment (16x32 bf16): row = lane%16, one contiguous 16-elem run:
//   k = kbase + 16h + [0..15]
static __device__ __forceinline__ v16bf ldA(const __bf16* row, int kbase, int h) {
  u4x2 t;
  t.a = *(const uint4*)(row + kbase + 16 * h);
  t.b = *(const uint4*)(row + kbase + 16 * h + 8);
  return __builtin_bit_cast(v16bf, t);
}
// B fragment (32x16 bf16): col = lane%16, one contiguous 16-elem run: k = kbase + 16h + [0..15]
static __device__ __forceinline__ v16bf ldB(const __bf16* p) {
  u4x2 t;
  t.a = *(const uint4*)(p);
  t.b = *(const uint4*)(p + 8);
  return __builtin_bit_cast(v16bf, t);
}

static __device__ __forceinline__ unsigned int packbf2(float a, float b) {
  unsigned short ux = __builtin_bit_cast(unsigned short, (__bf16)a);
  unsigned short uy = __builtin_bit_cast(unsigned short, (__bf16)b);
  return (unsigned int)ux | ((unsigned int)uy << 16);
}
static __device__ __forceinline__ uint4 pack8(const v8f v, float mul) {
  uint4 u;
  u.x = packbf2(v[0] * mul, v[1] * mul);
  u.y = packbf2(v[2] * mul, v[3] * mul);
  u.z = packbf2(v[4] * mul, v[5] * mul);
  u.w = packbf2(v[6] * mul, v[7] * mul);
  return u;
}

// async copy of one 16-byte segment: global -> LDS (ASYNCcnt-tracked, no VGPR round trip)
static __device__ __forceinline__ void async_b128(unsigned lds_byte_addr, const void* gaddr) {
  asm volatile("global_load_async_to_lds_b128 %0, %1, off"
               :: "v"(lds_byte_addr), "v"(gaddr) : "memory");
}

// =================== Kernel 1: fused QKV projection (transposed) ===================
// {Q,K}^T = W^T x X^T -> stored [tok][ch]; V stored transposed VT[ch][tok]
__global__ __launch_bounds__(128) void proj_qkv(
    const float* __restrict__ x,
    const float* __restrict__ Wa, const float* __restrict__ ba,
    const float* __restrict__ Wb, const float* __restrict__ bb,
    const float* __restrict__ Wc, const float* __restrict__ bc,
    __bf16* __restrict__ Qo, __bf16* __restrict__ Ko, __bf16* __restrict__ VT)
{
  __shared__ __bf16 Wt[3][CH][C_IN];     // transposed weights [outch][k], 48 KB

  const int tid = threadIdx.x;
  const int rowbase = blockIdx.x * 64;

  // stage W^T: pack two consecutive k per dword store
  const float* Wsrc[3] = {Wa, Wb, Wc};
  for (int p = 0; p < 3; ++p) {
    for (int i = tid; i < (C_IN / 2) * CH; i += 128) {
      int k2 = i >> 6, c = i & 63;     // k2: 0..63, c: 0..63
      float f0 = Wsrc[p][(2 * k2) * CH + c];
      float f1 = Wsrc[p][(2 * k2 + 1) * CH + c];
      ((unsigned int*)&Wt[p][c][0])[k2] = packbf2(f0, f1);
    }
  }
  __syncthreads();

  const int wave = tid >> 5;
  const int lane = tid & 31;
  const int h  = lane >> 4;
  const int ln = lane & 15;
  const int tok = rowbase + wave * 16 + ln;   // this lane's token (B-matrix column)

  // B fragments: x row of this token, 128 channels -> 4 fragments (f32 -> bf16)
  v16bf xb[4];
#pragma unroll
  for (int kb = 0; kb < 4; ++kb) {
    const float* xp = x + (size_t)tok * C_IN + kb * 32 + 16 * h;
    float4 f0 = *(const float4*)(xp);
    float4 f1 = *(const float4*)(xp + 4);
    float4 f2 = *(const float4*)(xp + 8);
    float4 f3 = *(const float4*)(xp + 12);
    float ftmp[16];
    *(float4*)&ftmp[0]  = f0; *(float4*)&ftmp[4]  = f1;
    *(float4*)&ftmp[8]  = f2; *(float4*)&ftmp[12] = f3;
#pragma unroll
    for (int e = 0; e < 16; ++e) xb[kb][e] = (__bf16)ftmp[e];
  }

  const float* bias[3] = {ba, bb, bc};
#pragma unroll
  for (int p = 0; p < 3; ++p) {
#pragma unroll
    for (int t = 0; t < 4; ++t) {
      // bias along D rows (out-channels t*16 + 8h + r)
      float4 b0 = *(const float4*)(bias[p] + t * 16 + 8 * h);
      float4 b1 = *(const float4*)(bias[p] + t * 16 + 8 * h + 4);
      v8f acc;
      acc[0] = b0.x; acc[1] = b0.y; acc[2] = b0.z; acc[3] = b0.w;
      acc[4] = b1.x; acc[5] = b1.y; acc[6] = b1.z; acc[7] = b1.w;
#pragma unroll
      for (int kb = 0; kb < 4; ++kb) {
        v16bf a = ldA(&Wt[p][t * 16 + ln][0], kb * 32, h);
        acc = __builtin_amdgcn_wmma_f32_16x16x32_bf16(false, a, false, xb[kb],
                                                      (short)0, acc, false, false);
      }
      if (p == 0) {
        *(uint4*)(Qo + (size_t)tok * CH + t * 16 + 8 * h) = pack8(acc, 1.0f);
      } else if (p == 1) {
        *(uint4*)(Ko + (size_t)tok * CH + t * 16 + 8 * h) = pack8(acc, 1.0f);
      } else {
        // V stored transposed: VT[ch][tok]; coalesced across lanes (tok = ln-contiguous)
#pragma unroll
        for (int r = 0; r < 8; ++r)
          VT[(size_t)(t * 16 + 8 * h + r) * NTOK + tok] = (__bf16)acc[r];
      }
    }
  }
}

// =================== Kernel 2: flash attention, double-buffered async staging ===
// S^T = K x Q^T ; O^T = V^T x P^T ; K and V^T chunks DMA'd to LDS via
// global_load_async_to_lds_b128 (ASYNCcnt), prefetching chunk i+1 under chunk i.
__global__ __launch_bounds__(256) void flash_attn(
    const __bf16* __restrict__ Q, const __bf16* __restrict__ Kb,
    const __bf16* __restrict__ VT, __bf16* __restrict__ ABC)
{
  __shared__ __bf16 Ks[2][64][CH];   // [key][ch]   2 x 8 KB
  __shared__ __bf16 Vs[2][CH][64];   // [ch][key]   2 x 8 KB

  const int tid  = threadIdx.x;
  const int wave = tid >> 5;
  const int lane = tid & 31;
  const int h  = lane >> 4;
  const int ln = lane & 15;
  const int qtok = blockIdx.x * 128 + wave * 16 + ln;  // this lane's query token

  const unsigned ldsK[2] = {(unsigned)(size_t)&Ks[0][0][0], (unsigned)(size_t)&Ks[1][0][0]};
  const unsigned ldsV[2] = {(unsigned)(size_t)&Vs[0][0][0], (unsigned)(size_t)&Vs[1][0][0]};

  // issue one chunk's staging: 2 K segs + 2 V segs per thread (4 async ops / wave)
  auto issue_chunk = [&](int chunk, int bufsel) {
    const __bf16* kg = Kb + (size_t)chunk * 64 * CH;   // 8 KB contiguous
    const unsigned lk = ldsK[bufsel], lv = ldsV[bufsel];
#pragma unroll
    for (int it = 0; it < 2; ++it) {
      int s = tid + it * 256;                           // 0..511
      async_b128(lk + (unsigned)s * 16u, (const void*)(kg + s * 8));
    }
#pragma unroll
    for (int it = 0; it < 2; ++it) {
      int s = tid + it * 256;
      int ch = s >> 3, part = s & 7;
      async_b128(lv + (unsigned)(ch * 64 + part * 8) * 2u,
                 (const void*)(VT + (size_t)ch * NTOK + (size_t)chunk * 64 + part * 8));
    }
  };

  // Q as B fragments (k = channel, col = query): contiguous from global, loaded once
  v16bf qb[2];
#pragma unroll
  for (int cc = 0; cc < 2; ++cc)
    qb[cc] = ldB(Q + (size_t)qtok * CH + cc * 32 + 16 * h);

  v8f o[4];
#pragma unroll
  for (int t = 0; t < 4; ++t)
#pragma unroll
    for (int r = 0; r < 8; ++r) o[t][r] = 0.f;
  float m = -3.0e38f, l = 0.f;

  const int NCHUNK = NTOK / 64;
  issue_chunk(0, 0);

  for (int ci = 0; ci < NCHUNK; ++ci) {
    const int buf = ci & 1;
    __syncthreads();                     // all consumers of buf^1 (chunk ci-1) done
    const int nxt = (ci + 1 < NCHUNK) ? (ci + 1) : 0;
    issue_chunk(nxt, buf ^ 1);           // prefetch under this chunk's compute
    asm volatile("s_wait_asynccnt 0x4" ::: "memory");  // my chunk-ci loads landed
    __syncthreads();                     // everyone's chunk-ci loads landed

    const __bf16 (*Kc)[CH] = Ks[buf];
    const __bf16 (*Vc)[64] = Vs[buf];

    // S^T tiles (rows = keys, cols = queries): 4 tiles x 2 wmma
    v8f s[4];
#pragma unroll
    for (int kt = 0; kt < 4; ++kt) {
      v8f z;
#pragma unroll
      for (int r = 0; r < 8; ++r) z[r] = 0.f;
#pragma unroll
      for (int cc = 0; cc < 2; ++cc) {
        v16bf a = ldA(&Kc[kt * 16 + ln][0], cc * 32, h);
        z = __builtin_amdgcn_wmma_f32_16x16x32_bf16(false, a, false, qb[cc],
                                                    (short)0, z, false, false);
      }
      s[kt] = z;
    }

    // online softmax: keys for query ln live in-lane (+ the partner half-lane)
    float cm = -3.0e38f;
#pragma unroll
    for (int kt = 0; kt < 4; ++kt)
#pragma unroll
      for (int r = 0; r < 8; ++r) cm = fmaxf(cm, s[kt][r]);
    cm = fmaxf(cm, __shfl_xor(cm, 16, 32));
    float mnew  = fmaxf(m, cm);
    float scale = __expf(m - mnew);
    float rs = 0.f;
#pragma unroll
    for (int kt = 0; kt < 4; ++kt)
#pragma unroll
      for (int r = 0; r < 8; ++r) {
        float p = __expf(s[kt][r] - mnew);
        s[kt][r] = p;
        rs += p;
      }
    rs += __shfl_xor(rs, 16, 32);
    l = l * scale + rs;
    m = mnew;
#pragma unroll
    for (int t = 0; t < 4; ++t)
#pragma unroll
      for (int r = 0; r < 8; ++r) o[t][r] *= scale;

    // build P^T B-fragments in registers (cross-half exchange, no LDS bounce)
    v16bf pb[2];
#pragma unroll
    for (int kb2 = 0; kb2 < 2; ++kb2) {
#pragma unroll
      for (int r = 0; r < 8; ++r) {
        float t0 = s[2 * kb2][r];
        float t1 = s[2 * kb2 + 1][r];
        float w0 = __shfl_xor(t0, 16, 32);
        float w1 = __shfl_xor(t1, 16, 32);
        pb[kb2][r]     = (__bf16)(h ? w1 : t0);
        pb[kb2][r + 8] = (__bf16)(h ? t1 : w0);
      }
    }

    // O^T += V^T x P^T : A from transposed V (contiguous keys)
#pragma unroll
    for (int kb2 = 0; kb2 < 2; ++kb2) {
#pragma unroll
      for (int t = 0; t < 4; ++t) {
        v16bf a = ldA(&Vc[t * 16 + ln][0], kb2 * 32, h);
        o[t] = __builtin_amdgcn_wmma_f32_16x16x32_bf16(false, a, false, pb[kb2],
                                                       (short)0, o[t], false, false);
      }
    }
  }

  // normalize (per-query scalar) and store: lane holds 8 consecutive channels
  float inv = 1.0f / l;
#pragma unroll
  for (int t = 0; t < 4; ++t)
    *(uint4*)(ABC + (size_t)qtok * CH + t * 16 + 8 * h) = pack8(o[t], inv);
}

// =================== Kernel 3: output projection + residual (transposed) ===================
// out^T = Wo^T x ABC^T : A = Wo^T (LDS), B = ABC rows (global, contiguous)
__global__ __launch_bounds__(128) void proj_out(
    const __bf16* __restrict__ ABC, const float* __restrict__ Wo,
    const float* __restrict__ bo, const float* __restrict__ x,
    float* __restrict__ out)
{
  __shared__ __bf16 Wot[C_IN][CH];       // [outch][k] 16 KB

  const int tid = threadIdx.x;
  const int rowbase = blockIdx.x * 64;

  // stage Wo^T: Wo is [64][128]; pack two consecutive k per dword
  for (int i = tid; i < (CH / 2) * C_IN; i += 128) {
    int k2 = i & 31, c = i >> 5;       // k2: 0..31, c: 0..127
    float f0 = Wo[(2 * k2) * C_IN + c];
    float f1 = Wo[(2 * k2 + 1) * C_IN + c];
    ((unsigned int*)&Wot[c][0])[k2] = packbf2(f0, f1);
  }
  __syncthreads();

  const int wave = tid >> 5;
  const int lane = tid & 31;
  const int h  = lane >> 4;
  const int ln = lane & 15;
  const int tok = rowbase + wave * 16 + ln;

  // B fragments: ABC row of this token (k = C2 channels), contiguous from global
  v16bf bb[2];
#pragma unroll
  for (int kc = 0; kc < 2; ++kc)
    bb[kc] = ldB(ABC + (size_t)tok * CH + kc * 32 + 16 * h);

#pragma unroll
  for (int t = 0; t < 8; ++t) {
    float4 b0 = *(const float4*)(bo + t * 16 + 8 * h);
    float4 b1 = *(const float4*)(bo + t * 16 + 8 * h + 4);
    v8f acc;
    acc[0] = b0.x; acc[1] = b0.y; acc[2] = b0.z; acc[3] = b0.w;
    acc[4] = b1.x; acc[5] = b1.y; acc[6] = b1.z; acc[7] = b1.w;
#pragma unroll
    for (int kc = 0; kc < 2; ++kc) {
      v16bf a = ldA(&Wot[t * 16 + ln][0], kc * 32, h);
      acc = __builtin_amdgcn_wmma_f32_16x16x32_bf16(false, a, false, bb[kc],
                                                    (short)0, acc, false, false);
    }
    // residual + store: 8 consecutive out-channels of this token
    float* op = out + (size_t)tok * C_IN + t * 16 + 8 * h;
    const float* xp = x + (size_t)tok * C_IN + t * 16 + 8 * h;
    float4 x0 = *(const float4*)(xp);
    float4 x1 = *(const float4*)(xp + 4);
    float4 r0, r1;
    r0.x = acc[0] + x0.x; r0.y = acc[1] + x0.y; r0.z = acc[2] + x0.z; r0.w = acc[3] + x0.w;
    r1.x = acc[4] + x1.x; r1.y = acc[5] + x1.y; r1.z = acc[6] + x1.z; r1.w = acc[7] + x1.w;
    *(float4*)(op)     = r0;
    *(float4*)(op + 4) = r1;
  }
}

extern "C" void kernel_launch(void* const* d_in, const int* in_sizes, int n_in,
                              void* d_out, int out_size, void* d_ws, size_t ws_size,
                              hipStream_t stream) {
  (void)in_sizes; (void)n_in; (void)out_size; (void)ws_size;
  const float* x  = (const float*)d_in[0];
  const float* Wa = (const float*)d_in[1];
  const float* ba = (const float*)d_in[2];
  const float* Wb = (const float*)d_in[3];
  const float* bb = (const float*)d_in[4];
  const float* Wc = (const float*)d_in[5];
  const float* bc = (const float*)d_in[6];
  const float* Wo = (const float*)d_in[7];
  const float* bo = (const float*)d_in[8];
  float* out = (float*)d_out;

  char* ws = (char*)d_ws;
  const size_t seg = (size_t)NTOK * CH * sizeof(__bf16);   // 2 MB each
  __bf16* Q   = (__bf16*)(ws);
  __bf16* Kb  = (__bf16*)(ws + seg);
  __bf16* VT  = (__bf16*)(ws + 2 * seg);   // V transposed: [CH][NTOK]
  __bf16* ABC = (__bf16*)(ws + 3 * seg);

  proj_qkv<<<NTOK / 64, 128, 0, stream>>>(x, Wa, ba, Wb, bb, Wc, bc, Q, Kb, VT);
  flash_attn<<<NTOK / 128, 256, 0, stream>>>(Q, Kb, VT, ABC);
  proj_out<<<NTOK / 64, 128, 0, stream>>>(ABC, Wo, bo, x, out);
}
